// HLSQoREstimator_13537736917037
// MI455X (gfx1250) — matrix operand order repaired and checked by reading, not compiled, not run on hardware
//
#include <hip/hip_runtime.h>

typedef unsigned short u16;
typedef unsigned int   u32;

typedef __attribute__((ext_vector_type(16))) __bf16 v16bf;
typedef __attribute__((ext_vector_type(8)))  float  v8f;

union FragB { v16bf v; u32 u[8]; };
union FragC { v8f  v; float f[8]; };

__device__ inline u32 packbf2(float a, float b) {
  u32 ua = __float_as_uint(a), ub = __float_as_uint(b);
  u32 ra = (ua + 0x7FFFu + ((ua >> 16) & 1u)) >> 16;
  u32 rb = (ub + 0x7FFFu + ((ub >> 16) & 1u)) >> 16;
  return (ra & 0xFFFFu) | (rb << 16);
}
__device__ inline u16 f2bf(float a) {
  u32 u = __float_as_uint(a);
  return (u16)((u + 0x7FFFu + ((u >> 16) & 1u)) >> 16);
}
__device__ inline float bf2f(u16 h) { return __uint_as_float(((u32)h) << 16); }
__device__ inline float bflo(u32 u) { return __uint_as_float(u << 16); }
__device__ inline float bfhi(u32 u) { return __uint_as_float(u & 0xFFFF0000u); }
__device__ inline float geluf(float x) { return 0.5f * x * (1.0f + erff(x * 0.7071067811865475f)); }
__device__ inline u32 fkey(float f) { u32 u = __float_as_uint(f); return (u & 0x80000000u) ? ~u : (u | 0x80000000u); }
__device__ inline float fdec(u32 k) { return __uint_as_float((k & 0x80000000u) ? (k & 0x7FFFFFFFu) : ~k); }

#define GF_GELU   1
#define GF_SKIP   2
#define GF_OUTB16 4

// One-time weight re-pack: W[K x 128] f32 -> WMMA B-fragment-layout bf16 image.
// Output is exactly the LDS image consumed per k-step (2048 dwords per k-step).
__global__ void k_packw(const float* __restrict__ W, u32* __restrict__ WP, int K) {
  int t = blockIdx.x * blockDim.x + threadIdx.x;
  int total = (K >> 5) << 11;
  if (t >= total) return;
  int ks  = t >> 11;
  int idx = t & 2047;
  int nt = idx >> 8;
  int ln = (idx >> 3) & 31;
  int v  = idx & 7;
  int c  = nt * 16 + (ln & 15);
  int kk = ks * 32 + ((v >> 2) << 4) + ((ln >> 4) << 3) + 2 * (v & 3);
  WP[t] = packbf2(W[(size_t)kk * 128 + c], W[(size_t)(kk + 1) * 128 + c]);
}

// out[M x 128] = act(A[M x K]) @ W[K x 128] + bias, W pre-packed (WP).
// Weight k-step chunk staged to LDS with global_load_async_to_lds_b128 (ASYNCcnt),
// overlapped with the A-tile f32 loads; inner loop = 16x ds_read_b128 + 8x wmma.
__global__ __launch_bounds__(256) void k_gemm(
    const float* __restrict__ A, int lda, int aoff, int M,
    const u32* __restrict__ WP, const float* __restrict__ bias, int K,
    float* __restrict__ outF, u16* __restrict__ outH, int ldo, int ooff,
    int flags, const float* __restrict__ skipX, int skipLda, int skipOff,
    const float* __restrict__ gatePtr)
{
  __shared__ u32 wl[2048];
  const int tid  = threadIdx.x;
  const int wave = tid >> 5, lane = tid & 31;
  const int col16 = lane & 15, khalf = lane >> 4;
  const int tile = blockIdx.x * 8 + wave;
  const bool active = (tile * 16) < M;
  const int arow = tile * 16 + col16;
  const unsigned ldsA = (unsigned)(size_t)(&wl[tid * 8]);  // low 32 bits = LDS byte offset

  FragC acc[8];
#pragma unroll
  for (int t = 0; t < 8; ++t)
#pragma unroll
    for (int r = 0; r < 8; ++r) acc[t].f[r] = 0.0f;

  for (int kb = 0; kb < K; kb += 32) {
    __syncthreads();   // all waves done reading previous chunk
    // async DMA: 32 bytes per thread of the pre-packed weight image -> LDS
    const u32* gsrc = WP + (((size_t)kb >> 5) << 11) + tid * 8;
    asm volatile(
        "global_load_async_to_lds_b128 %0, %1, off\n\t"
        "global_load_async_to_lds_b128 %0, %1, off offset:16"
        :: "v"(ldsA), "v"(gsrc) : "memory");

    FragB afr;
    if (active) {  // overlap A loads (+ optional gelu + bf16 pack) with the DMA
      const float* ap = A + (size_t)arow * lda + aoff + kb + khalf * 8;
#pragma unroll
      for (int j = 0; j < 4; ++j) {
        float x0 = ap[2 * j],      x1 = ap[2 * j + 1];
        float y0 = ap[16 + 2 * j], y1 = ap[17 + 2 * j];
        if (flags & GF_GELU) { x0 = geluf(x0); x1 = geluf(x1); y0 = geluf(y0); y1 = geluf(y1); }
        afr.u[j]     = packbf2(x0, x1);
        afr.u[4 + j] = packbf2(y0, y1);
      }
    }
    asm volatile("s_wait_asynccnt 0" ::: "memory");
    __syncthreads();

    if (active) {
      FragB bf[8];
#pragma unroll
      for (int t = 0; t < 8; ++t) {
        const u32* bp = &wl[(t << 8) + (lane << 3)];
#pragma unroll
        for (int j = 0; j < 8; ++j) bf[t].u[j] = bp[j];
      }
#pragma unroll
      for (int t = 0; t < 8; ++t)
        acc[t].v = __builtin_amdgcn_wmma_f32_16x16x32_bf16(
            false, afr.v, false, bf[t].v, (short)0, acc[t].v, false, false);
    }
  }

  if (active) {
    float g = 1.0f, gi = 0.0f;
    if (flags & GF_SKIP) {
      float s = *gatePtr;
      g = 1.0f / (1.0f + expf(-s));
      gi = 1.0f - g;
    }
#pragma unroll
    for (int t = 0; t < 8; ++t) {
      int c = t * 16 + col16;
      float b = bias ? bias[c] : 0.0f;
#pragma unroll
      for (int r = 0; r < 8; ++r) {
        int rr = tile * 16 + r + (khalf << 3);
        float val = acc[t].f[r] + b;
        if (flags & GF_SKIP)
          val = g * val + gi * skipX[(size_t)rr * skipLda + skipOff + c];
        if (flags & GF_OUTB16) outH[(size_t)rr * ldo + ooff + c] = f2bf(val);
        else                   outF[(size_t)rr * ldo + ooff + c] = val;
      }
    }
  }
}

// KREL = K @ blockdiag(a_rel[h]); VREL = V @ blockdiag(m_rel[h]) (bf16 in/out)
__global__ __launch_bounds__(256) void k_rel(
    const u16* __restrict__ Kin, const u16* __restrict__ Vin,
    const float* __restrict__ Arel, const float* __restrict__ Mrel,
    u16* __restrict__ KrelO, u16* __restrict__ VrelO, int Ns)
{
  __shared__ u32 wa[2048];
  __shared__ u32 wm[2048];
  const int tid = threadIdx.x, wave = tid >> 5, lane = tid & 31;
  const int col16 = lane & 15, khalf = lane >> 4;

  for (int idx = tid; idx < 2048; idx += 256) {
    int h  = idx >> 9;
    int t2 = (idx >> 8) & 1;
    int ln = (idx >> 3) & 31;
    int v  = idx & 7;
    int c  = t2 * 16 + (ln & 15);
    int kk = ((v >> 2) << 4) + ((ln >> 4) << 3) + 2 * (v & 3);
    wa[idx] = packbf2(Arel[h * 1024 + kk * 32 + c], Arel[h * 1024 + (kk + 1) * 32 + c]);
    wm[idx] = packbf2(Mrel[h * 1024 + kk * 32 + c], Mrel[h * 1024 + (kk + 1) * 32 + c]);
  }
  __syncthreads();

  int tiles = Ns >> 4;
  for (int tile = blockIdx.x * 8 + wave; tile < tiles; tile += gridDim.x * 8) {
    int arow = tile * 16 + col16;
    const u16* kp = Kin + (size_t)arow * 128;
    const u16* vp = Vin + (size_t)arow * 128;
#pragma unroll
    for (int h = 0; h < 4; ++h) {
      FragB ak, av;
      uint4 q0 = *(const uint4*)(kp + h * 32 + khalf * 8);
      uint4 q1 = *(const uint4*)(kp + h * 32 + 16 + khalf * 8);
      ak.u[0] = q0.x; ak.u[1] = q0.y; ak.u[2] = q0.z; ak.u[3] = q0.w;
      ak.u[4] = q1.x; ak.u[5] = q1.y; ak.u[6] = q1.z; ak.u[7] = q1.w;
      uint4 r0 = *(const uint4*)(vp + h * 32 + khalf * 8);
      uint4 r1 = *(const uint4*)(vp + h * 32 + 16 + khalf * 8);
      av.u[0] = r0.x; av.u[1] = r0.y; av.u[2] = r0.z; av.u[3] = r0.w;
      av.u[4] = r1.x; av.u[5] = r1.y; av.u[6] = r1.z; av.u[7] = r1.w;
#pragma unroll
      for (int t2 = 0; t2 < 2; ++t2) {
        FragB bk, bm;
        const u32* pk = &wa[((h * 2 + t2) << 8) + (lane << 3)];
        const u32* pm = &wm[((h * 2 + t2) << 8) + (lane << 3)];
#pragma unroll
        for (int j = 0; j < 8; ++j) { bk.u[j] = pk[j]; bm.u[j] = pm[j]; }
        FragC ck, cv;
#pragma unroll
        for (int r = 0; r < 8; ++r) { ck.f[r] = 0.0f; cv.f[r] = 0.0f; }
        ck.v = __builtin_amdgcn_wmma_f32_16x16x32_bf16(false, ak.v, false, bk.v, (short)0, ck.v, false, false);
        cv.v = __builtin_amdgcn_wmma_f32_16x16x32_bf16(false, av.v, false, bm.v, (short)0, cv.v, false, false);
        int cbase = h * 32 + t2 * 16 + col16;
#pragma unroll
        for (int r = 0; r < 8; ++r) {
          int rr = tile * 16 + r + (khalf << 3);
          KrelO[(size_t)rr * 128 + cbase] = f2bf(ck.f[r]);
          VrelO[(size_t)rr * 128 + cbase] = f2bf(cv.f[r]);
        }
      }
    }
  }
}

__global__ void k_alpha(const int* __restrict__ src, const int* __restrict__ dst, int E,
                        const u16* __restrict__ Q, const u16* __restrict__ Krel,
                        const float* __restrict__ prel,
                        float* __restrict__ alpha, u32* __restrict__ amaxB)
{
  int t = blockIdx.x * blockDim.x + threadIdx.x;
  int e = t >> 2, h = t & 3;
  if (e >= E) return;
  int s = src[e], d = dst[e];
  const u16* qp = Q    + (size_t)d * 128 + h * 32;
  const u16* kp = Krel + (size_t)s * 128 + h * 32;
  float sum = 0.0f;
#pragma unroll
  for (int j = 0; j < 32; j += 8) {
    uint4 a = *(const uint4*)(qp + j);
    uint4 b = *(const uint4*)(kp + j);
    sum += bflo(a.x) * bflo(b.x) + bfhi(a.x) * bfhi(b.x);
    sum += bflo(a.y) * bflo(b.y) + bfhi(a.y) * bfhi(b.y);
    sum += bflo(a.z) * bflo(b.z) + bfhi(a.z) * bfhi(b.z);
    sum += bflo(a.w) * bflo(b.w) + bfhi(a.w) * bfhi(b.w);
  }
  float al = sum * prel[h] * 0.17677669529663687f;
  alpha[(size_t)e * 4 + h] = al;
  atomicMax(&amaxB[(size_t)d * 4 + h], fkey(al));
}

__global__ void k_fill_u32(u32* p, u32 v, int n) {
  int t = blockIdx.x * blockDim.x + threadIdx.x;
  if (t < n) p[t] = v;
}

__global__ void k_amax_fin(const u32* __restrict__ amaxB, float* __restrict__ amaxF, int n) {
  int t = blockIdx.x * blockDim.x + threadIdx.x;
  if (t >= n) return;
  float f = fdec(amaxB[t]);
  if (!(fabsf(f) < 3.0e38f)) f = 0.0f;
  amaxF[t] = f;
}

__global__ void k_exp(const int* __restrict__ dst, int E,
                      float* __restrict__ alpha,
                      const float* __restrict__ amaxF, float* __restrict__ den)
{
  int t = blockIdx.x * blockDim.x + threadIdx.x;
  int e = t >> 2, h = t & 3;
  if (e >= E) return;
  int d = dst[e];
  float ex = expf(alpha[(size_t)e * 4 + h] - amaxF[(size_t)d * 4 + h]);
  alpha[(size_t)e * 4 + h] = ex;
  atomicAdd(&den[(size_t)d * 4 + h], ex);
}

__global__ void k_agg(const int* __restrict__ src, const int* __restrict__ dst, int E,
                      const float* __restrict__ alpha, const float* __restrict__ den,
                      const u16* __restrict__ Vrel, float* __restrict__ agg)
{
  int t = blockIdx.x * blockDim.x + threadIdx.x;
  int e = t >> 5, lane = t & 31;
  if (e >= E) return;
  int s = src[e], d = dst[e];
#pragma unroll
  for (int q = 0; q < 4; ++q) {
    int j = q * 32 + lane;
    float att = alpha[(size_t)e * 4 + q] / (den[(size_t)d * 4 + q] + 1e-16f);
    float v = bf2f(Vrel[(size_t)s * 128 + j]);
    atomicAdd(&agg[(size_t)d * 128 + j], att * v);
  }
}

__global__ void k_count(const int* __restrict__ batch, int N, float* __restrict__ cnt) {
  int t = blockIdx.x * blockDim.x + threadIdx.x;
  if (t < N) atomicAdd(&cnt[batch[t]], 1.0f);
}

__global__ void k_ln_stats(const float* __restrict__ X, int lda, int off, int N,
                           const int* __restrict__ batch,
                           float* __restrict__ S, float* __restrict__ Q2)
{
  int t = blockIdx.x * blockDim.x + threadIdx.x;
  int n = t >> 5, lane = t & 31;
  if (n >= N) return;
  const float* row = X + (size_t)n * lda + off;
  float s = 0.0f, q = 0.0f;
  for (int j = lane; j < 128; j += 32) { float x = row[j]; s += x; q += x * x; }
  for (int d = 16; d > 0; d >>= 1) { s += __shfl_down(s, d, 32); q += __shfl_down(q, d, 32); }
  if (lane == 0) { int b = batch[n]; atomicAdd(&S[b], s); atomicAdd(&Q2[b], q); }
}

__global__ void k_ln_fin(const float* __restrict__ S, const float* __restrict__ Q2,
                         const float* __restrict__ cnt, int B,
                         float* __restrict__ mean, float* __restrict__ inv)
{
  int t = blockIdx.x * blockDim.x + threadIdx.x;
  if (t >= B) return;
  float norm = fmaxf(cnt[t], 1.0f) * 128.0f;
  float m = S[t] / norm;
  float v = Q2[t] / norm - m * m;
  mean[t] = m;
  inv[t] = rsqrtf(v + 1e-5f);
}

__global__ void k_ln_apply(float* __restrict__ X, int lda, int off, int N,
                           const int* __restrict__ batch,
                           const float* __restrict__ mean, const float* __restrict__ inv,
                           const float* __restrict__ w, const float* __restrict__ b)
{
  int t = blockIdx.x * blockDim.x + threadIdx.x;
  if (t >= N * 128) return;
  int n = t >> 7, c = t & 127;
  int g = batch[n];
  float* p = X + (size_t)n * lda + off + c;
  *p = (*p - mean[g]) * inv[g] * w[c] + b[c];
}

__global__ void k_pool(const float* __restrict__ X, int N, const int* __restrict__ batch,
                       float* __restrict__ H, int base, u32* __restrict__ xxB)
{
  int t = blockIdx.x * blockDim.x + threadIdx.x;
  int n = t >> 5, lane = t & 31;
  if (n >= N) return;
  int b = batch[n];
#pragma unroll
  for (int q = 0; q < 4; ++q) {
    int j = q * 32 + lane;
    float x = X[(size_t)n * 128 + j];
    atomicAdd(&H[(size_t)b * 1552 + base + j], x);
    atomicMax(&xxB[b * 128 + j], fkey(x));
  }
}

__global__ void k_pool_fin(float* __restrict__ H, int base, const u32* __restrict__ xxB,
                           const float* __restrict__ cnt)
{
  int t = blockIdx.x * blockDim.x + threadIdx.x;
  if (t >= 64 * 128) return;
  int b = t >> 7, c = t & 127;
  float xa = H[(size_t)b * 1552 + base + c];
  H[(size_t)b * 1552 + base + 128 + c] = xa / fmaxf(cnt[b], 1.0f);
  H[(size_t)b * 1552 + base + 256 + c] = fdec(xxB[b * 128 + c]);
}

__global__ void k_ymlp(const float* __restrict__ yb,
                       const float* __restrict__ W1, const float* __restrict__ b1,
                       const float* __restrict__ W2, const float* __restrict__ b2,
                       float* __restrict__ H)
{
  int t = blockIdx.x * blockDim.x + threadIdx.x;
  if (t >= 64) return;
  float y = yb[t];
  float h1[16];
#pragma unroll
  for (int j = 0; j < 16; ++j) h1[j] = geluf(y * W1[j] + b1[j]);
#pragma unroll
  for (int o = 0; o < 16; ++o) {
    float a = b2[o];
#pragma unroll
    for (int j = 0; j < 16; ++j) a += h1[j] * W2[j * 16 + o];
    H[(size_t)t * 1552 + 1536 + o] = a;
  }
}

__global__ void k_mlp_gemm(const float* __restrict__ A, const float* __restrict__ W,
                           const float* __restrict__ bias, float* __restrict__ O,
                           int M, int K, int Nout)
{
  int t = blockIdx.x * blockDim.x + threadIdx.x;
  if (t >= M * Nout) return;
  int r = t / Nout, o = t - r * Nout;
  float acc = bias[o];
  const float* a = A + (size_t)r * K;
  for (int k = 0; k < K; ++k) acc += a[k] * W[(size_t)k * Nout + o];
  O[t] = acc;
}

__global__ __launch_bounds__(256) void k_ln_gelu_row(float* __restrict__ X, int F,
                                                     const float* __restrict__ w,
                                                     const float* __restrict__ b)
{
  __shared__ float ss[8], sq[8];
  int row = blockIdx.x;
  float s = 0.0f, q = 0.0f;
  for (int j = threadIdx.x; j < F; j += blockDim.x) {
    float x = X[(size_t)row * F + j];
    s += x; q += x * x;
  }
  for (int d = 16; d > 0; d >>= 1) { s += __shfl_down(s, d, 32); q += __shfl_down(q, d, 32); }
  int wave = threadIdx.x >> 5, lane = threadIdx.x & 31;
  if (lane == 0) { ss[wave] = s; sq[wave] = q; }
  __syncthreads();
  if (threadIdx.x == 0) {
    float S = 0.0f, Q = 0.0f;
    for (int i = 0; i < 8; ++i) { S += ss[i]; Q += sq[i]; }
    float m = S / F;
    ss[0] = m;
    sq[0] = Q / F - m * m;
  }
  __syncthreads();
  float m = ss[0], inv = rsqrtf(sq[0] + 1e-5f);
  for (int j = threadIdx.x; j < F; j += blockDim.x) {
    float x = X[(size_t)row * F + j];
    X[(size_t)row * F + j] = geluf((x - m) * inv * w[j] + b[j]);
  }
}

__global__ void k_final(const float* __restrict__ A, const float* __restrict__ W,
                        const float* __restrict__ b, float* __restrict__ out)
{
  int t = blockIdx.x * blockDim.x + threadIdx.x;
  if (t >= 64) return;
  float acc = b[0];
  for (int k = 0; k < 128; ++k) acc += A[(size_t)t * 128 + k] * W[k];
  out[t] = acc;
}

extern "C" void kernel_launch(void* const* d_in, const int* in_sizes, int n_in,
                              void* d_out, int out_size, void* d_ws, size_t ws_size,
                              hipStream_t stream)
{
  (void)in_sizes; (void)n_in; (void)out_size; (void)ws_size;

  const int NN[4]   = {100000, 50000, 20000, 10000};   // instr, var, const, block
  const int ROFF[4] = {0, 100000, 150000, 170000};
  const int BIDX[4] = {11, 12, 13, 14};
  const int ALPH[4] = {2, 3, 1, 0};
  const int PROJ[4] = {1, 2, 0, -1};
  const int TOTN = 180000;

  struct ETI { int s, d, E, ei, rel, roff, aoff; };
  const ETI ETS[7] = {
    {0, 0, 200000, 4, 3, 0,      0},        // flow
    {1, 0, 150000, 5, 6, 100000, 200000},   // use
    {0, 1,  80000, 6, 2, 150000, 350000},   // def
    {2, 0,  80000, 7, 1, 250000, 430000},   // cuse
    {0, 2,  50000, 8, 5, 270000, 510000},   // val
    {3, 0,  80000, 9, 0, 370000, 560000},   // contains
    {0, 3,  80000, 10, 4, 380000, 640000},  // to
  };

  const int PB = 16;   // params start after 16 data inputs; JAX pytree leaf order
  auto jkW  = [&](int nt)        { return PB + 2 * ALPH[nt]; };
  auto LBv  = [&](int l)         { return PB + 8 + 57 * l; };
  auto aW   = [&](int l, int nt) { return LBv(l) + 0  + 2 * ALPH[nt]; };
  auto arel = [&](int l, int r)  { return LBv(l) + 8  + r; };
  auto kW   = [&](int l, int nt) { return LBv(l) + 15 + 2 * ALPH[nt]; };
  auto mrel = [&](int l, int r)  { return LBv(l) + 23 + r; };
  auto prelI= [&](int l, int r)  { return LBv(l) + 30 + r; };
  auto qW   = [&](int l, int nt) { return LBv(l) + 37 + 2 * ALPH[nt]; };
  auto skpI = [&](int l, int nt) { return LBv(l) + 45 + ALPH[nt]; };
  auto vW   = [&](int l, int nt) { return LBv(l) + 49 + 2 * ALPH[nt]; };
  auto mlpI = [&](int j)         { return PB + 179 + j; };
  auto nrmI = [&](int i, int nt, int wb) { return PB + 193 + 8 * i + 2 * ALPH[nt] + wb; };
  auto prjI = [&](int nt)        { return PB + 209 + 2 * PROJ[nt]; };
  const int YMLP = PB + 215;

  auto PF = [&](int i) -> const float* { return (const float*)d_in[i]; };
  auto PI = [&](int i) -> const int*   { return (const int*)d_in[i];   };

  char* wsp = (char*)d_ws;
  size_t woff = 0;
  auto alloc = [&](size_t bytes) -> char* {
    char* p = wsp + woff;
    woff = (woff + bytes + 255) & ~(size_t)255;
    return p;
  };
  float* X0    = (float*)alloc((size_t)TOTN * 128 * 4);
  float* XS    = (float*)alloc((size_t)TOTN * 384 * 4);
  u16*   Kb    = (u16*)  alloc((size_t)TOTN * 128 * 2);
  u16*   Qb    = (u16*)  alloc((size_t)TOTN * 128 * 2);
  u16*   Vb    = (u16*)  alloc((size_t)TOTN * 128 * 2);
  u16*   KREL  = (u16*)  alloc((size_t)480000 * 128 * 2);
  u16*   VREL  = (u16*)  alloc((size_t)480000 * 128 * 2);
  float* ALP   = (float*)alloc((size_t)720000 * 4 * 4);
  u32*   AMB   = (u32*)  alloc((size_t)TOTN * 4 * 4);
  float* AMF   = (float*)alloc((size_t)TOTN * 4 * 4);
  float* DEN   = (float*)alloc((size_t)TOTN * 4 * 4);
  float* AGG   = (float*)alloc((size_t)TOTN * 128 * 4);
  float* CNT   = (float*)alloc(4 * 64 * 4);
  float* SB    = (float*)alloc(64 * 4);
  float* QB2   = (float*)alloc(64 * 4);
  float* MEANB = (float*)alloc(64 * 4);
  float* INVB  = (float*)alloc(64 * 4);
  u32*   XXB   = (u32*)  alloc(64 * 128 * 4);
  float* H     = (float*)alloc((size_t)64 * 1552 * 4);
  float* M1    = (float*)alloc((size_t)64 * 512 * 4);
  float* M2    = (float*)alloc((size_t)64 * 256 * 4);
  float* M3    = (float*)alloc((size_t)64 * 128 * 4);
  u32*   WPB   = (u32*)  alloc((size_t)503808 * 4);   // packed-weight arena

  // ---- one-time weight packing into WMMA B-frag bf16 images ----
  u32* wp_cur = WPB;
  auto packw = [&](const float* W, int K) -> const u32* {
    u32* dst = wp_cur;
    int total = (K >> 5) << 11;
    wp_cur += total;
    k_packw<<<(total + 255) / 256, 256, 0, stream>>>(W, dst, K);
    return dst;
  };
  const u32* WPproj[3];
  for (int nt = 0; nt < 3; ++nt) WPproj[nt] = packw(PF(prjI(nt)), 64);
  const u32 *WPk[3][4], *WPq[3][4], *WPv[3][4], *WPa[3][4];
  for (int l = 0; l < 3; ++l)
    for (int nt = 0; nt < 4; ++nt) {
      WPk[l][nt] = packw(PF(kW(l, nt)), 128);
      WPq[l][nt] = packw(PF(qW(l, nt)), 128);
      WPv[l][nt] = packw(PF(vW(l, nt)), 128);
      WPa[l][nt] = packw(PF(aW(l, nt)), 128);
    }
  const u32* WPjk[4];
  for (int nt = 0; nt < 4; ++nt) WPjk[nt] = packw(PF(jkW(nt)), 384);

  auto gemm = [&](const float* A, int lda, int aoff, int M,
                  const u32* WP, const float* bias, int K,
                  float* oF, u16* oH, int ldo, int ooff, int flags,
                  const float* sX, int sLda, int sOff, const float* gate) {
    int blocks = (M / 16 + 7) / 8;
    k_gemm<<<dim3(blocks), dim3(256), 0, stream>>>(A, lda, aoff, M, WP, bias, K,
                                                   oF, oH, ldo, ooff, flags, sX, sLda, sOff, gate);
  };

  // input projection (block passes through)
  hipMemcpyAsync(X0 + (size_t)ROFF[3] * 128, d_in[3], (size_t)NN[3] * 128 * 4,
                 hipMemcpyDeviceToDevice, stream);
  for (int nt = 0; nt < 3; ++nt)
    gemm(PF(nt), 64, 0, NN[nt], WPproj[nt], PF(prjI(nt) + 1), 64,
         X0 + (size_t)ROFF[nt] * 128, nullptr, 128, 0, 0, nullptr, 0, 0, nullptr);

  hipMemsetAsync(CNT, 0, 4 * 64 * 4, stream);
  for (int nt = 0; nt < 4; ++nt)
    k_count<<<(NN[nt] + 255) / 256, 256, 0, stream>>>(PI(BIDX[nt]), NN[nt], CNT + nt * 64);

  for (int l = 0; l < 3; ++l) {
    for (int nt = 0; nt < 4; ++nt) {
      const float* Xin = (l == 0) ? X0 + (size_t)ROFF[nt] * 128 : XS + (size_t)ROFF[nt] * 384;
      int lda  = (l == 0) ? 128 : 384;
      int aoff = (l == 0) ? 0 : (l - 1) * 128;
      gemm(Xin, lda, aoff, NN[nt], WPk[l][nt], PF(kW(l, nt) + 1), 128,
           nullptr, Kb + (size_t)ROFF[nt] * 128, 128, 0, GF_OUTB16, nullptr, 0, 0, nullptr);
      gemm(Xin, lda, aoff, NN[nt], WPq[l][nt], PF(qW(l, nt) + 1), 128,
           nullptr, Qb + (size_t)ROFF[nt] * 128, 128, 0, GF_OUTB16, nullptr, 0, 0, nullptr);
      gemm(Xin, lda, aoff, NN[nt], WPv[l][nt], PF(vW(l, nt) + 1), 128,
           nullptr, Vb + (size_t)ROFF[nt] * 128, 128, 0, GF_OUTB16, nullptr, 0, 0, nullptr);
    }
    for (int e = 0; e < 7; ++e) {
      if (l == 0 && e == 5) continue;
      const ETI& et = ETS[e];
      int blocks = (NN[et.s] / 16 + 7) / 8;
      k_rel<<<dim3(blocks), dim3(256), 0, stream>>>(
          Kb + (size_t)ROFF[et.s] * 128, Vb + (size_t)ROFF[et.s] * 128,
          PF(arel(l, et.rel)), PF(mrel(l, et.rel)),
          KREL + (size_t)et.roff * 128, VREL + (size_t)et.roff * 128, NN[et.s]);
    }
    k_fill_u32<<<(TOTN * 4 + 255) / 256, 256, 0, stream>>>(AMB, 0x007FFFFFu, TOTN * 4);
    hipMemsetAsync(DEN, 0, (size_t)TOTN * 4 * 4, stream);
    hipMemsetAsync(AGG, 0, (size_t)TOTN * 128 * 4, stream);
    for (int e = 0; e < 7; ++e) {
      if (l == 0 && e == 5) continue;
      const ETI& et = ETS[e];
      k_alpha<<<(et.E * 4 + 255) / 256, 256, 0, stream>>>(
          PI(et.ei), PI(et.ei) + et.E, et.E,
          Qb + (size_t)ROFF[et.d] * 128, KREL + (size_t)et.roff * 128,
          PF(prelI(l, et.rel)), ALP + (size_t)et.aoff * 4, AMB + (size_t)ROFF[et.d] * 4);
    }
    k_amax_fin<<<(TOTN * 4 + 255) / 256, 256, 0, stream>>>(AMB, AMF, TOTN * 4);
    for (int e = 0; e < 7; ++e) {
      if (l == 0 && e == 5) continue;
      const ETI& et = ETS[e];
      k_exp<<<(et.E * 4 + 255) / 256, 256, 0, stream>>>(
          PI(et.ei) + et.E, et.E, ALP + (size_t)et.aoff * 4,
          AMF + (size_t)ROFF[et.d] * 4, DEN + (size_t)ROFF[et.d] * 4);
    }
    for (int e = 0; e < 7; ++e) {
      if (l == 0 && e == 5) continue;
      const ETI& et = ETS[e];
      k_agg<<<(et.E * 32 + 255) / 256, 256, 0, stream>>>(
          PI(et.ei), PI(et.ei) + et.E, et.E,
          ALP + (size_t)et.aoff * 4, DEN + (size_t)ROFF[et.d] * 4,
          VREL + (size_t)et.roff * 128, AGG + (size_t)ROFF[et.d] * 128);
    }
    for (int nt = 0; nt < 4; ++nt) {
      const float* Xin = (l == 0) ? X0 + (size_t)ROFF[nt] * 128 : XS + (size_t)ROFF[nt] * 384;
      int lda  = (l == 0) ? 128 : 384;
      int aoff = (l == 0) ? 0 : (l - 1) * 128;
      gemm(AGG + (size_t)ROFF[nt] * 128, 128, 0, NN[nt],
           WPa[l][nt], PF(aW(l, nt) + 1), 128,
           XS + (size_t)ROFF[nt] * 384, nullptr, 384, l * 128,
           GF_GELU | GF_SKIP, Xin, lda, aoff, PF(skpI(l, nt)));
    }
    if (l < 2) {
      for (int nt = 0; nt < 4; ++nt) {
        hipMemsetAsync(SB, 0, 64 * 4, stream);
        hipMemsetAsync(QB2, 0, 64 * 4, stream);
        k_ln_stats<<<(NN[nt] * 32 + 255) / 256, 256, 0, stream>>>(
            XS + (size_t)ROFF[nt] * 384, 384, l * 128, NN[nt], PI(BIDX[nt]), SB, QB2);
        k_ln_fin<<<1, 64, 0, stream>>>(SB, QB2, CNT + nt * 64, 64, MEANB, INVB);
        k_ln_apply<<<(NN[nt] * 128 + 255) / 256, 256, 0, stream>>>(
            XS + (size_t)ROFF[nt] * 384, 384, l * 128, NN[nt], PI(BIDX[nt]),
            MEANB, INVB, PF(nrmI(l, nt, 0)), PF(nrmI(l, nt, 1)));
      }
    }
  }

  // jumping-knowledge projection
  for (int nt = 0; nt < 4; ++nt)
    gemm(XS + (size_t)ROFF[nt] * 384, 384, 0, NN[nt],
         WPjk[nt], PF(jkW(nt) + 1), 384,
         AGG + (size_t)ROFF[nt] * 128, nullptr, 128, 0, 0, nullptr, 0, 0, nullptr);

  // pooling
  hipMemsetAsync(H, 0, (size_t)64 * 1552 * 4, stream);
  for (int nt = 0; nt < 4; ++nt) {
    k_fill_u32<<<(64 * 128 + 255) / 256, 256, 0, stream>>>(XXB, 0x007FFFFFu, 64 * 128);
    k_pool<<<(NN[nt] * 32 + 255) / 256, 256, 0, stream>>>(
        AGG + (size_t)ROFF[nt] * 128, NN[nt], PI(BIDX[nt]), H, nt * 384, XXB);
    k_pool_fin<<<(64 * 128 + 255) / 256, 256, 0, stream>>>(H, nt * 384, XXB, CNT + nt * 64);
  }

  // y_base MLP + head
  k_ymlp<<<1, 64, 0, stream>>>(PF(15), PF(YMLP), PF(YMLP + 1), PF(YMLP + 2), PF(YMLP + 3), H);
  k_mlp_gemm<<<(64 * 512 + 255) / 256, 256, 0, stream>>>(H, PF(mlpI(0)), PF(mlpI(1)), M1, 64, 1552, 512);
  k_ln_gelu_row<<<64, 256, 0, stream>>>(M1, 512, PF(mlpI(2)), PF(mlpI(3)));
  k_mlp_gemm<<<(64 * 256 + 255) / 256, 256, 0, stream>>>(M1, PF(mlpI(4)), PF(mlpI(5)), M2, 64, 512, 256);
  k_ln_gelu_row<<<64, 256, 0, stream>>>(M2, 256, PF(mlpI(6)), PF(mlpI(7)));
  k_mlp_gemm<<<(64 * 128 + 255) / 256, 256, 0, stream>>>(M2, PF(mlpI(8)), PF(mlpI(9)), M3, 64, 256, 128);
  k_ln_gelu_row<<<64, 256, 0, stream>>>(M3, 128, PF(mlpI(10)), PF(mlpI(11)));
  k_final<<<1, 64, 0, stream>>>(M3, PF(mlpI(12)), PF(mlpI(13)), (float*)d_out);
}
